// drugcell_nn_knockout_64115271794769
// MI455X (gfx1250) — matrix operand order repaired and checked by reading, not compiled, not run on hardware
//
#include <hip/hip_runtime.h>
#include <hip/hip_bf16.h>
#include <stdint.h>

// ---------------------------------------------------------------------------
// DrugCell-style network for MI455X (gfx1250, wave32, WMMA).
// Heavy stages use v_wmma_f32_16x16x32_f16 with LDS-staged B chunks in a
// 3-deep software pipeline: global->reg load, reg->LDS store, and LDS->WMMA
// consume each live in separate barrier-delimited regions, so staging loads
// always have a full half-iteration of latency slack.
// ---------------------------------------------------------------------------

typedef _Float16 v16h __attribute__((ext_vector_type(16)));
typedef _Float16 v8h  __attribute__((ext_vector_type(8)));
typedef float    v8f  __attribute__((ext_vector_type(8)));

#define G_    3008
#define D_    2048
#define B_    8
#define E_    128
#define XW_   5057       // G + D + 1
#define NT_   11         // number of hierarchy terms
#define ND_   704        // 11 * 64 direct outputs
#define KCH_  94         // 3008 / 32 K-chunks (even -> unroll by 2)
#define MT1_  188        // 3008 / 16 M-tiles (stage 1)
#define MT2_  64         // 1024 / 16 M-tiles (stage 2)

struct Ptr11      { const float* p[11]; };
struct TermParams {
  const float* lin_w[11];  const float* lin_b[11];
  const float* gate_w[11]; const float* gate_b[11];
  const float* a1w[11];    const float* a1b[11];
  const float* a2w[11];    const float* a2b[11];
};

static __device__ __forceinline__ v8f wmma_f16(v16h a, v16h b, v8f c) {
  return __builtin_amdgcn_wmma_f32_16x16x32_f16(false, a, false, b, (short)0, c,
                                                false, false);
}

// ---------------------------------------------------------------------------
// K1: masked gene vector: gene[b,g] = x[b,g] * table[didx[b],g], knockouts -> 0
// ---------------------------------------------------------------------------
__global__ void k_gene(const float* __restrict__ x, const float* __restrict__ table,
                       const int* __restrict__ ko, float* __restrict__ gene) {
  int idx = blockIdx.x * 256 + threadIdx.x;
  if (idx >= B_ * G_) return;
  int b = idx / G_, g = idx - b * G_;
  int didx = (int)x[b * XW_ + G_ + D_];
  float v = x[b * XW_ + g];
  if (g == ko[0] || g == ko[1] || g == ko[2] || g == ko[3]) v = 0.f;
  v *= table[didx * G_ + g];
  gene[idx] = v;
}

// ---------------------------------------------------------------------------
// K2: pre-swizzle rbf weights into WMMA B-fragment order (f16).
// B layout (16-bit, 32x16): N=lane&15; K=8*(lane>>4)+h (h<8), 16+8*(lane>>4)+(h-8) (h>=8).
// w16sw[((kb*8 + t)*32 + lane)*16 + h] = w[e=t*16+(lane&15)][j=kb*32+K]
// ---------------------------------------------------------------------------
__global__ void k_w16sw(const float* __restrict__ rbf_w, _Float16* __restrict__ w16sw) {
  int idx = blockIdx.x * 256 + threadIdx.x;
  if (idx >= KCH_ * 8 * 32 * 16) return;
  int h    = idx & 15;
  int lane = (idx >> 4) & 31;
  int t    = (idx >> 9) & 7;
  int kb   = idx >> 12;
  int e = t * 16 + (lane & 15);
  int j = kb * 32 + ((h < 8) ? 8 * (lane >> 4) + h : 16 + 8 * (lane >> 4) + (h - 8));
  w16sw[idx] = (_Float16)rbf_w[e * G_ + j];
}

// ---------------------------------------------------------------------------
// K3: pre-swizzle the 11 direct_t weights (concatenated, 3008 x 704) likewise.
// ---------------------------------------------------------------------------
__global__ void k_wd16sw(Ptr11 dw, _Float16* __restrict__ wd16sw) {
  int idx = blockIdx.x * 256 + threadIdx.x;
  if (idx >= KCH_ * 44 * 32 * 16) return;
  int h    = idx & 15;
  int lane = (idx >> 4) & 31;
  int q    = idx >> 9;
  int t    = q % 44;
  int kb   = q / 44;
  int n    = t * 16 + (lane & 15);
  int term = n >> 6, c = n & 63;
  int j = kb * 32 + ((h < 8) ? 8 * (lane >> 4) + h : 16 + 8 * (lane >> 4) + (h - 8));
  wd16sw[idx] = (_Float16)dw.p[term][c * G_ + j];
}

// ---------------------------------------------------------------------------
// K4: fused RBF + GEMM.  vect[b,i,e] = sum_j exp(-0.5*(gi-gj)^2) * w[e,j] + be
// One wave = 16-row x 128-col strip; 47 blocks x 4 waves = 188 tiles, no
// divergence. Per half-iteration: load chunk k+2 to regs, store chunk k+1
// regs to LDS, consume chunk k from LDS (exp A-fragment + 8 WMMAs), barrier.
// Output stored transposed as f16: gperm16[(b*128+e)*3008 + i].
// ---------------------------------------------------------------------------
__device__ __forceinline__ void rbf_chunk_lds(const float* gs, float gi, int hi,
                                              int kb, const v16h* blds,
                                              v8f (&acc)[8]) {
  union { v16h v; _Float16 h[16]; } A;
  int base = kb * 32 + (hi ? 8 : 0);
  #pragma unroll
  for (int hh = 0; hh < 16; ++hh) {           // A 16x32: K = hh (<8) / hh+8 (>=8)
    float d = gi - gs[base + (hh < 8 ? hh : hh + 8)];
    A.h[hh] = (_Float16)__expf(-0.5f * d * d);
  }
  #pragma unroll
  for (int t = 0; t < 8; ++t) acc[t] = wmma_f16(A.v, blds[t * 32], acc[t]);
}

__global__ void __launch_bounds__(128)
k_rbf_gemm(const float* __restrict__ gene, const _Float16* __restrict__ w16sw,
           const float* __restrict__ rbf_b, _Float16* __restrict__ gperm16) {
  __shared__ float gs[G_];
  __shared__ __align__(32) uint4 Bsh[2][512];   // 2 x 8KB B-chunk double buffer
  int tid = threadIdx.x;                        // 128 threads = 4 waves
  int b = blockIdx.y;
  for (int i = tid; i < G_; i += 128) gs[i] = gene[b * G_ + i];

  const uint4* bsrc = (const uint4*)w16sw;      // chunk kb at bsrc + kb*512
  {                                             // stage chunk 0 (64B/thread)
    #pragma unroll
    for (int q = 0; q < 4; ++q) Bsh[0][tid * 4 + q] = bsrc[tid * 4 + q];
  }
  __syncthreads();

  int wave = tid >> 5, lane = tid & 31;
  int tile = blockIdx.x * 4 + wave;             // 47 blocks x 4 waves = 188
  int i0 = tile * 16;
  int mrow = lane & 15, hi = lane >> 4;
  float gi = gs[i0 + mrow];

  v8f acc[8];
  #pragma unroll
  for (int t = 0; t < 8; ++t)
    #pragma unroll
    for (int r = 0; r < 8; ++r) acc[t][r] = 0.0f;

  uint4 S0, S1, S2, S3;                         // staged chunk (regs), odd
  {
    const uint4* s = bsrc + 512 + tid * 4;      // chunk 1
    S0 = s[0]; S1 = s[1]; S2 = s[2]; S3 = s[3];
  }

  for (int kb = 0; kb < KCH_; kb += 2) {
    uint4 T0, T1, T2, T3;                       // staged chunk (regs), even
    {   // half A: load k+2, store S (k+1) -> Bsh[1], consume Bsh[0] (k)
      int k2 = (kb + 2 < KCH_) ? kb + 2 : 0;    // tail prefetch redundant
      const uint4* s = bsrc + (size_t)k2 * 512 + tid * 4;
      T0 = s[0]; T1 = s[1]; T2 = s[2]; T3 = s[3];
      uint4* d = &Bsh[1][tid * 4];
      d[0] = S0; d[1] = S1; d[2] = S2; d[3] = S3;
      rbf_chunk_lds(gs, gi, hi, kb, (const v16h*)&Bsh[0][0] + lane, acc);
    }
    __syncthreads();
    {   // half B: load k+3, store T (k+2) -> Bsh[0], consume Bsh[1] (k+1)
      int k3 = (kb + 3 < KCH_) ? kb + 3 : 0;
      const uint4* s = bsrc + (size_t)k3 * 512 + tid * 4;
      S0 = s[0]; S1 = s[1]; S2 = s[2]; S3 = s[3];
      uint4* d = &Bsh[0][tid * 4];
      d[0] = T0; d[1] = T1; d[2] = T2; d[3] = T3;
      rbf_chunk_lds(gs, gi, hi, kb + 1, (const v16h*)&Bsh[1][0] + lane, acc);
    }
    __syncthreads();
  }

  int rbase = i0 + hi * 8;                      // D rows: M = r + 8*hi
  #pragma unroll
  for (int t = 0; t < 8; ++t) {
    int e = t * 16 + mrow;                      // D cols: N = lane&15
    float bias = rbf_b[e];
    union { v8h v; _Float16 h[8]; } o;
    #pragma unroll
    for (int r = 0; r < 8; ++r) o.h[r] = (_Float16)(acc[t][r] + bias);
    *(v8h*)(gperm16 + (size_t)(b * E_ + e) * G_ + rbase) = o.v;  // 16B aligned
  }
}

// ---------------------------------------------------------------------------
// K5: direct linears as one GEMM: (1024 x 3008) @ (3008 x 704) + bias.
// 88 blocks = 11 term-groups x 8 blocks: all 8 waves share one ng; 4KB B
// chunks use the same 3-deep pipeline; per-wave A stream double buffered.
// ---------------------------------------------------------------------------
__device__ __forceinline__ void dir_load_a(const _Float16* arow, int kb, int hi,
                                           v16h& A) {
  int base = kb * 32 + hi * 8;
  union { v16h v; v8h pr[2]; } t;
  t.pr[0] = *(const v8h*)(arow + base);        // K = base..base+7
  t.pr[1] = *(const v8h*)(arow + base + 16);   // K = base+16..base+23
  A = t.v;
}

__device__ __forceinline__ void wmma4_lds(v16h A, const v16h* bl, v8f (&acc)[4]) {
  #pragma unroll
  for (int t = 0; t < 4; ++t) acc[t] = wmma_f16(A, bl[t * 32], acc[t]);
}

__global__ void __launch_bounds__(256)
k_direct_gemm(const _Float16* __restrict__ gperm16, const _Float16* __restrict__ wd16sw,
              Ptr11 db, float* __restrict__ direct_out) {
  __shared__ __align__(32) uint4 Bsh[2][256];   // 2 x 4KB B-chunk double buffer
  int tid = threadIdx.x;
  int wave = tid >> 5, lane = tid & 31;
  int ng = blockIdx.x >> 3;                     // term group, shared by block
  int mt = (blockIdx.x & 7) * 8 + wave;         // M tile, per wave
  int ng4 = ng * 4;
  int i0 = mt * 16;
  int mrow = lane & 15, hi = lane >> 4;
  const _Float16* arow = gperm16 + (size_t)(i0 + mrow) * G_;
  const uint4* bsrc = (const uint4*)wd16sw;     // chunk kb at (kb*44+ng4)*64

  v16h A0, A1;
  dir_load_a(arow, 0, hi, A0);
  Bsh[0][tid] = bsrc[(size_t)ng4 * 64 + tid];   // stage chunk 0
  __syncthreads();

  v8f acc[4];
  #pragma unroll
  for (int t = 0; t < 4; ++t)
    #pragma unroll
    for (int r = 0; r < 8; ++r) acc[t][r] = 0.0f;

  uint4 S = bsrc[(size_t)(44 + ng4) * 64 + tid];          // chunk 1
  for (int kb = 0; kb < KCH_; kb += 2) {
    uint4 T;
    {   // half A: load k+2, store S (k+1) -> Bsh[1], consume Bsh[0]/A0 (k)
      int k2 = (kb + 2 < KCH_) ? kb + 2 : 0;    // tail prefetch redundant
      T = bsrc[(size_t)(k2 * 44 + ng4) * 64 + tid];
      dir_load_a(arow, kb + 1, hi, A1);
      Bsh[1][tid] = S;
      wmma4_lds(A0, (const v16h*)&Bsh[0][0] + lane, acc);
    }
    __syncthreads();
    {   // half B: load k+3, store T (k+2) -> Bsh[0], consume Bsh[1]/A1 (k+1)
      int k2 = (kb + 2 < KCH_) ? kb + 2 : 0;
      int k3 = (kb + 3 < KCH_) ? kb + 3 : 0;
      S = bsrc[(size_t)(k3 * 44 + ng4) * 64 + tid];
      dir_load_a(arow, k2, hi, A0);
      Bsh[0][tid] = T;
      wmma4_lds(A1, (const v16h*)&Bsh[1][0] + lane, acc);
    }
    __syncthreads();
  }

  int rbase = i0 + hi * 8;
  #pragma unroll
  for (int t = 0; t < 4; ++t) {
    int c = t * 16 + mrow;
    int n = ng * 64 + c;
    float bias = db.p[ng][c];
    #pragma unroll
    for (int r = 0; r < 8; ++r)
      direct_out[(size_t)(rbase + r) * ND_ + n] = acc[t][r] + bias;
  }
}

// ---------------------------------------------------------------------------
// Small scalar stages
// ---------------------------------------------------------------------------
__device__ inline void bn8(float* v) {      // batchnorm over 8 batches, eps 1e-5
  float m = 0.f;
  for (int b = 0; b < 8; ++b) m += v[b];
  m *= 0.125f;
  float var = 0.f;
  for (int b = 0; b < 8; ++b) { float d = v[b] - m; var += d * d; }
  var *= 0.125f;
  float inv = rsqrtf(var + 1e-5f);
  for (int b = 0; b < 8; ++b) v[b] = (v[b] - m) * inv;
}

__global__ void __launch_bounds__(256)
k_drug(const float* __restrict__ x,
       const float* w0, const float* b0, const float* w1, const float* b1,
       const float* w2, const float* b2, float* __restrict__ drug_out) {
  __shared__ float zb[100 * 8];
  __shared__ float za[50 * 8];
  __shared__ float zc[6 * 8];
  int tid = threadIdx.x;
  for (int id = tid; id < 800; id += 256) {            // 2048 -> 100
    int o = id >> 3, b = id & 7;
    const float* xin = x + b * XW_ + G_;
    const float* wr = w0 + o * D_;
    float s = b0[o];
    for (int j = 0; j < D_; ++j) s += wr[j] * xin[j];
    zb[o * 8 + b] = tanhf(s);
  }
  __syncthreads();
  if (tid < 100) bn8(&zb[tid * 8]);
  __syncthreads();
  for (int id = tid; id < 400; id += 256) {            // 100 -> 50
    int o = id >> 3, b = id & 7;
    const float* wr = w1 + o * 100;
    float s = b1[o];
    for (int j = 0; j < 100; ++j) s += wr[j] * zb[j * 8 + b];
    za[o * 8 + b] = tanhf(s);
  }
  __syncthreads();
  if (tid < 50) bn8(&za[tid * 8]);
  __syncthreads();
  if (tid < 48) {                                       // 50 -> 6
    int o = tid >> 3, b = tid & 7;
    const float* wr = w2 + o * 50;
    float s = b2[o];
    for (int j = 0; j < 50; ++j) s += wr[j] * za[j * 8 + b];
    zc[o * 8 + b] = tanhf(s);
  }
  __syncthreads();
  if (tid < 6) bn8(&zc[tid * 8]);
  __syncthreads();
  if (tid < 48) {
    int o = tid >> 3, b = tid & 7;
    drug_out[b * 6 + o] = zc[o * 8 + b];
  }
}

// One block per hierarchy term. kind: 0=leaf, 1=mid, 2=root.
// hbuf layout: [c*1024 + b*128 + p]; termout layout: [((term*8+b)*16+c)*128+p]
__global__ void __launch_bounds__(256)
k_term(int kind, TermParams tp, const float* __restrict__ direct_out,
       const float* __restrict__ drug_out, float* __restrict__ termout,
       float* __restrict__ hbuf_all, float* __restrict__ aux) {
  int term, nk, insz, kid0;
  if (kind == 0)      { term = blockIdx.x;     nk = 0; insz = 64;  kid0 = 0; }
  else if (kind == 1) { term = 8 + blockIdx.x; nk = 4; insz = 128; kid0 = blockIdx.x * 4; }
  else                { term = 10;             nk = 2; insz = 96;  kid0 = 8; }

  __shared__ float hbm[B_ * E_];
  __shared__ float stats[32];
  __shared__ float gateS[8];
  int tid = threadIdx.x;
  float* hbuf = hbuf_all + (size_t)term * 16384;
  const float* lw = tp.lin_w[term];
  const float* lb = tp.lin_b[term];

  for (int id = tid; id < 16384; id += 256) {           // h = lin(ci), tanh
    int c = id >> 10, rem = id & 1023, b = rem >> 7, p = rem & 127;
    const float* wr = lw + c * insz;
    float s = lb[c];
    int k = 0;
    for (; k < nk * 16; ++k) {
      int kid = kid0 + (k >> 4), cc = k & 15;
      s += wr[k] * termout[(((size_t)kid * 8 + b) * 16 + cc) * 128 + p];
    }
    const float* dp = direct_out + (size_t)(b * 128 + p) * ND_ + term * 64;
    for (int kk = 0; k < insz; ++k, ++kk) s += wr[k] * dp[kk];
    hbuf[id] = tanhf(s);
  }
  __syncthreads();
  if (tid < 16) {                                       // BN stats over (b,p)
    const float* hv = hbuf + tid * 1024;
    float m = 0.f;
    for (int i = 0; i < 1024; ++i) m += hv[i];
    m *= (1.f / 1024.f);
    float var = 0.f;
    for (int i = 0; i < 1024; ++i) { float d = hv[i] - m; var += d * d; }
    var *= (1.f / 1024.f);
    stats[tid * 2] = m; stats[tid * 2 + 1] = rsqrtf(var + 1e-5f);
  }
  __syncthreads();
  for (int id = tid; id < 16384; id += 256) {
    int c = id >> 10;
    hbuf[id] = (hbuf[id] - stats[c * 2]) * stats[c * 2 + 1];
  }
  __syncthreads();
  for (int id = tid; id < 1024; id += 256) {            // mean over channels
    float s = 0.f;
    for (int c = 0; c < 16; ++c) s += hbuf[c * 1024 + id];
    hbm[id] = s * (1.f / 16.f);
  }
  __syncthreads();
  if (tid < 8) {                                        // gate
    int b = tid;
    const float* gw = tp.gate_w[term];
    float s = tp.gate_b[term][0];
    for (int p = 0; p < 128; ++p) s += gw[p] * hbm[b * 128 + p];
    for (int j = 0; j < 6; ++j)   s += gw[128 + j] * drug_out[b * 6 + j];
    gateS[b] = 1.f / (1.f + __expf(-s));
  }
  __syncthreads();
  for (int id = tid; id < 16384; id += 256) {           // term_out = gate * hb
    int c = id >> 10, rem = id & 1023, b = rem >> 7, p = rem & 127;
    termout[(((size_t)term * 8 + b) * 16 + c) * 128 + p] = gateS[b] * hbuf[id];
  }
  if (tid < 8) {                                        // aux head
    int b = tid;
    const float* a1 = tp.a1w[term];
    float s = tp.a1b[term][0];
    for (int p = 0; p < 128; ++p) s += a1[p] * gateS[b] * hbm[b * 128 + p];
    aux[term * 8 + b] = tp.a2w[term][0] * tanhf(s) + tp.a2b[term][0];
  }
}

__global__ void __launch_bounds__(256)
k_final(const float* __restrict__ termout, const float* __restrict__ drug_out,
        const float* fw, const float* fb, const float* faw, const float* fab,
        const float* fow, const float* fob, const float* __restrict__ aux,
        float* __restrict__ out) {
  __shared__ float rm[1024];
  __shared__ float z[6 * 8];
  int tid = threadIdx.x;
  for (int id = tid; id < 1024; id += 256) {            // root mean over ch
    int b = id >> 7, p = id & 127;
    float s = 0.f;
    for (int c = 0; c < 16; ++c)
      s += termout[(((size_t)10 * 8 + b) * 16 + c) * 128 + p];
    rm[id] = s * (1.f / 16.f);
  }
  __syncthreads();
  if (tid < 48) {                                       // final linear + tanh
    int o = tid >> 3, b = tid & 7;
    const float* wr = fw + o * 134;
    float s = fb[o];
    for (int p = 0; p < 128; ++p) s += wr[p] * rm[b * 128 + p];
    for (int j = 0; j < 6; ++j)   s += wr[128 + j] * drug_out[b * 6 + j];
    z[o * 8 + b] = tanhf(s);
  }
  __syncthreads();
  if (tid < 6) bn8(&z[tid * 8]);
  __syncthreads();
  if (tid < 8) {                                        // aux_final
    int b = tid;
    float s = fab[0];
    for (int o = 0; o < 6; ++o) s += faw[o] * z[o * 8 + b];
    out[b] = fow[0] * tanhf(s) + fob[0];
  }
  if (tid < 48) {                                       // out (8,6)
    int o = tid >> 3, b = tid & 7;
    out[8 + b * 6 + o] = z[o * 8 + b];
  }
  if (tid < 88) out[56 + tid] = aux[tid];               // aux_stack (11,8,1)
}

// ---------------------------------------------------------------------------
// Host side
// ---------------------------------------------------------------------------
extern "C" void kernel_launch(void* const* d_in, const int* in_sizes, int n_in,
                              void* d_out, int out_size, void* d_ws, size_t ws_size,
                              hipStream_t stream) {
  (void)n_in; (void)out_size; (void)ws_size;
  auto F = [&](int i) { return (const float*)d_in[i]; };

  const float *x, *table, *rbf_w, *rbf_b, *fw, *fb, *faw, *fab, *fow, *fob;
  const float *dw[3], *dbv[3];
  const int* ko;
  Ptr11 dirw, dirb;
  TermParams tp;

  // Two plausible pytree flattenings; discriminate via first input's size.
  if (in_sizes[0] == 4) {
    // JAX tree_leaves order: dict keys sorted at every level ('b' < 'w').
    ko = (const int*)d_in[0];
    for (int t = 0; t < 11; ++t) {
      int base = 1 + t * 8;   // <T>_aux1{b,w}, _aux2{b,w}, _gate{b,w}, _lin{b,w}
      tp.a1b[t] = F(base + 0); tp.a1w[t]  = F(base + 1);
      tp.a2b[t] = F(base + 2); tp.a2w[t]  = F(base + 3);
      tp.gate_b[t] = F(base + 4); tp.gate_w[t] = F(base + 5);
      tp.lin_b[t]  = F(base + 6); tp.lin_w[t]  = F(base + 7);
    }
    for (int t = 0; t < 11; ++t) { dirb.p[t] = F(89 + t * 2); dirw.p[t] = F(90 + t * 2); }
    for (int i = 0; i < 3; ++i)  { dbv[i] = F(111 + i * 2); dw[i] = F(112 + i * 2); }
    fb = F(117); fw = F(118); fab = F(119); faw = F(120); fob = F(121); fow = F(122);
    rbf_b = F(123); rbf_w = F(124); table = F(125); x = F(126);
  } else {
    // Insertion-order flatten ('w' before 'b' as built in _lin()).
    x = F(0); table = F(1); ko = (const int*)d_in[2];
    rbf_w = F(3); rbf_b = F(4);
    for (int i = 0; i < 3; ++i) { dw[i] = F(5 + i * 2); dbv[i] = F(6 + i * 2); }
    for (int t = 0; t < 11; ++t) {
      int base = 11 + t * 10;  // direct{w,b}, lin{w,b}, gate{w,b}, aux1{w,b}, aux2{w,b}
      dirw.p[t] = F(base + 0); dirb.p[t] = F(base + 1);
      tp.lin_w[t] = F(base + 2); tp.lin_b[t] = F(base + 3);
      tp.gate_w[t] = F(base + 4); tp.gate_b[t] = F(base + 5);
      tp.a1w[t] = F(base + 6); tp.a1b[t] = F(base + 7);
      tp.a2w[t] = F(base + 8); tp.a2b[t] = F(base + 9);
    }
    fw = F(121); fb = F(122); faw = F(123); fab = F(124); fow = F(125); fob = F(126);
  }

  // Workspace layout (bytes, 256-aligned); total ~15.6 MB
  char* ws = (char*)d_ws;
  float*    gene  = (float*)   (ws + 0);               //  96,256
  _Float16* w16   = (_Float16*)(ws + 96256);           // 770,048
  _Float16* wd16  = (_Float16*)(ws + 866304);          // 4,235,264
  _Float16* gp16  = (_Float16*)(ws + 5101568);         // 6,160,384
  float*    diro  = (float*)   (ws + 11261952);        // 2,883,584
  float*    drugo = (float*)   (ws + 14145536);        // 256
  float*    tout  = (float*)   (ws + 14145792);        // 720,896
  float*    hbuf  = (float*)   (ws + 14866688);        // 720,896
  float*    auxp  = (float*)   (ws + 15587584);        // 512

  k_gene  <<<94,   256, 0, stream>>>(x, table, ko, gene);
  k_w16sw <<<1504, 256, 0, stream>>>(rbf_w, w16);
  k_wd16sw<<<8272, 256, 0, stream>>>(dirw, wd16);
  k_rbf_gemm<<<dim3(47, 8), 128, 0, stream>>>(gene, w16, rbf_b, gp16);
  k_direct_gemm<<<88, 256, 0, stream>>>(gp16, wd16, dirb, diro);
  k_drug<<<1, 256, 0, stream>>>(x, dw[0], dbv[0], dw[1], dbv[1], dw[2], dbv[2], drugo);
  k_term<<<8, 256, 0, stream>>>(0, tp, diro, drugo, tout, hbuf, auxp);   // leaves
  k_term<<<2, 256, 0, stream>>>(1, tp, diro, drugo, tout, hbuf, auxp);   // mids
  k_term<<<1, 256, 0, stream>>>(2, tp, diro, drugo, tout, hbuf, auxp);   // root
  k_final<<<1, 256, 0, stream>>>(tout, drugo, fw, fb, faw, fab, fow, fob,
                                 auxp, (float*)d_out);
}
